// MOE_72722386255909
// MI455X (gfx1250) — matrix operand order repaired and checked by reading, not compiled, not run on hardware
//
#include <hip/hip_runtime.h>

// ---------------------------------------------------------------------------
// MoE transformer forward for MI455X (gfx1250, wave32, WMMA).
//
// Roofline: dominated by the 67 GF logits GEMM (2048x512 @ 512x32000); all
// weights fit in the 192MB L2, logits store (262MB) ~11us @ 23.3TB/s
// => matrix-core bound. Strategy:
//   * One-time f32->f16 conversion of every GEMM operand (weights ~50MB f16,
//     L2-resident) so the GEMM inner loop is pure global_load_b128 of f16
//     fragments + v_wmma_f32_16x16x32_f16 (f32 accumulate), no VALU converts.
//   * 32x64 tile per wave (2 A-frag x 4 B-frag -> 8 WMMA / 32-wide K step):
//     24 B loaded per WMMA per lane; halves L2 re-reads of the big out_w.
//   * Ping-pong double-buffered K loop (2 steps per iteration, buffers only
//     ever written by fresh loads -> zero v_mov rotation, max load->use gap).
//   * V pre-transposed to [KV,HD,T] f16 so P@V uses the contiguous B path.
// ---------------------------------------------------------------------------

#define BB   2
#define TT   1024
#define CC   512
#define HH   8
#define KVH  2
#define HDIM 64
#define LL   2
#define EE   8
#define VV   32000
#define NN   (BB * TT)   // 2048 tokens

typedef _Float16 v16h __attribute__((ext_vector_type(16)));
typedef _Float16 v8h  __attribute__((ext_vector_type(8)));
typedef float    v8f  __attribute__((ext_vector_type(8)));

// ---------------------------------------------------------------------------
// h[b,t,:] = tok_emb[ids[b,t],:] + pos_emb[t,:]
// ---------------------------------------------------------------------------
__global__ __launch_bounds__(256) void embed_kernel(
    const int* __restrict__ ids, const float* __restrict__ tok,
    const float* __restrict__ pos, float* __restrict__ h) {
  int i = blockIdx.x * 256 + threadIdx.x;
  if (i >= NN * CC) return;
  int c = i & (CC - 1);
  int n = i >> 9;            // / CC
  int t = n & (TT - 1);
  h[i] = tok[(size_t)ids[n] * CC + c] + pos[t * CC + c];
}

// ---------------------------------------------------------------------------
// f32 -> f16 bulk convert, 8 elements/thread (b128 in, b128 out).
// ---------------------------------------------------------------------------
__global__ __launch_bounds__(256) void cvt_kernel(
    const float* __restrict__ in, _Float16* __restrict__ out, int n8) {
  int i = blockIdx.x * 256 + threadIdx.x;
  if (i >= n8) return;
  float4 a = *(const float4*)(in + (size_t)i * 8);
  float4 b = *(const float4*)(in + (size_t)i * 8 + 4);
  v8h o;
  o[0] = (_Float16)a.x; o[1] = (_Float16)a.y;
  o[2] = (_Float16)a.z; o[3] = (_Float16)a.w;
  o[4] = (_Float16)b.x; o[5] = (_Float16)b.y;
  o[6] = (_Float16)b.z; o[7] = (_Float16)b.w;
  *(v8h*)(out + (size_t)i * 8) = o;
}

// ---------------------------------------------------------------------------
// V transpose+convert: vT[b][kv][d][t] = (f16) v[b][t][kv][d]
// Makes P@V's B operand contiguous along the contraction (keys) axis.
// ---------------------------------------------------------------------------
__global__ __launch_bounds__(256) void vtrans_kernel(
    const float* __restrict__ v, _Float16* __restrict__ vT) {
  int i = blockIdx.x * 256 + threadIdx.x;   // [b][kv][d][t], 2^18 total
  if (i >= BB * KVH * HDIM * TT) return;
  int t  = i & (TT - 1);
  int d  = (i >> 10) & (HDIM - 1);
  int kv = (i >> 16) & (KVH - 1);
  int b  = i >> 17;
  vT[i] = (_Float16)v[((size_t)(b * TT + t) * KVH + kv) * HDIM + d];
}

// ---------------------------------------------------------------------------
// RMSNorm: one block (256 threads) per row of C=512.
// ---------------------------------------------------------------------------
__global__ __launch_bounds__(256) void rmsnorm_kernel(
    const float* __restrict__ x, const float* __restrict__ w,
    float* __restrict__ out) {
  __shared__ float red[256];
  int n = blockIdx.x, tid = threadIdx.x;
  const float* xr = x + (size_t)n * CC;
  float a = xr[tid], b = xr[tid + 256];
  red[tid] = a * a + b * b;
  __syncthreads();
  for (int s = 128; s > 0; s >>= 1) {
    if (tid < s) red[tid] += red[tid + s];
    __syncthreads();
  }
  float rinv = rsqrtf(red[0] * (1.0f / CC) + 1e-6f);
  out[(size_t)n * CC + tid]       = a * rinv * w[tid];
  out[(size_t)n * CC + tid + 256] = b * rinv * w[tid + 256];
}

// ---------------------------------------------------------------------------
// Causal softmax: reads f32 score row, writes f16 probability row
// (next GEMM's A operand). scale = 1/sqrt(HD) = 0.125. Masked cols -> 0.
// ---------------------------------------------------------------------------
__global__ __launch_bounds__(256) void attn_softmax_kernel(
    const float* __restrict__ s, _Float16* __restrict__ p) {
  __shared__ float red[256];
  int row = blockIdx.x, tid = threadIdx.x;
  const float* sr = s + (size_t)row * TT;
  _Float16*    pr = p + (size_t)row * TT;
  const float scale = 0.125f;
  int nvalid = row + 1;
  float m = -1e30f;
  for (int j = tid; j < nvalid; j += 256) m = fmaxf(m, sr[j] * scale);
  red[tid] = m; __syncthreads();
  for (int st = 128; st > 0; st >>= 1) {
    if (tid < st) red[tid] = fmaxf(red[tid], red[tid + st]);
    __syncthreads();
  }
  m = red[0]; __syncthreads();
  float sum = 0.0f;
  for (int j = tid; j < nvalid; j += 256) sum += expf(sr[j] * scale - m);
  red[tid] = sum; __syncthreads();
  for (int st = 128; st > 0; st >>= 1) {
    if (tid < st) red[tid] += red[tid + st];
    __syncthreads();
  }
  float inv = 1.0f / red[0];
  for (int j = tid; j < TT; j += 256)
    pr[j] = (j < nvalid) ? (_Float16)(expf(sr[j] * scale - m) * inv)
                         : (_Float16)0.0f;
}

// ---------------------------------------------------------------------------
// MoE gating: logits = xn @ gate_w^T (E=8), top-2, softmax over top-2 logits,
// scatter into comb[n, E].
// ---------------------------------------------------------------------------
__global__ __launch_bounds__(256) void gate_top2_kernel(
    const float* __restrict__ xn, const float* __restrict__ gw,
    float* __restrict__ comb) {
  __shared__ float red[256];
  __shared__ float lg[EE];
  int n = blockIdx.x, tid = threadIdx.x;
  float x0 = xn[(size_t)n * CC + tid], x1 = xn[(size_t)n * CC + 256 + tid];
  for (int e = 0; e < EE; e++) {
    red[tid] = x0 * gw[e * CC + tid] + x1 * gw[e * CC + 256 + tid];
    __syncthreads();
    for (int s = 128; s > 0; s >>= 1) {
      if (tid < s) red[tid] += red[tid + s];
      __syncthreads();
    }
    if (tid == 0) lg[e] = red[0];
    __syncthreads();
  }
  if (tid == 0) {
    int i1 = 0;
    for (int e = 1; e < EE; e++) if (lg[e] > lg[i1]) i1 = e;   // top_k: first idx on tie
    int i2 = -1;
    for (int e = 0; e < EE; e++) {
      if (e == i1) continue;
      if (i2 < 0 || lg[e] > lg[i2]) i2 = e;
    }
    float eb = expf(lg[i2] - lg[i1]);
    float w0 = 1.0f / (1.0f + eb);
    for (int e = 0; e < EE; e++) comb[(size_t)n * EE + e] = 0.0f;
    comb[(size_t)n * EE + i1] = w0;
    comb[(size_t)n * EE + i2] = eb * w0;
  }
}

// ---------------------------------------------------------------------------
// Fragment loaders (wave32; contiguous k order k = 16h + s chosen here, not the ISA table order):
//  A 16x32 f16 : lane l = row l&15; here lanes<16 own K 0-15, lanes>=16 own
//                K 16-31 -> two contiguous 8-half (16B) loads.
//  B 32x16 f16 : lane n = column n&15; same order as A: lanes<16 own K 0-15, lanes>=16 own
//                K 16-31 -> one contiguous 16-half (32B) run, two 16B loads.
// ---------------------------------------------------------------------------
static __device__ __forceinline__ v16h load_a_frag(const _Float16* p) {
  v8h lo = *(const v8h*)(p);
  v8h hi = *(const v8h*)(p + 8);
  return __builtin_shufflevector(lo, hi, 0, 1, 2, 3, 4, 5, 6, 7,
                                         8, 9, 10, 11, 12, 13, 14, 15);
}
static __device__ __forceinline__ v16h load_b_frag(const _Float16* p) {
  v8h lo = *(const v8h*)(p);
  v8h hi = *(const v8h*)(p + 8);
  return __builtin_shufflevector(lo, hi, 0, 1, 2, 3, 4, 5, 6, 7,
                                         8, 9, 10, 11, 12, 13, 14, 15);
}
static __device__ __forceinline__ v8f wmma_f16(v16h a, v16h b, v8f c) {
  return __builtin_amdgcn_wmma_f32_16x16x32_f16(false, a, false, b,
                                                (short)0, c, false, false);
}

// 8 WMMAs for one 32x64 x K=32 step (2 row-groups x 4 col-groups)
#define WMMA8(A0, A1, B0, B1, B2, B3)                                         \
  acc[0] = wmma_f16(A0, B0, acc[0]); acc[1] = wmma_f16(A0, B1, acc[1]);       \
  acc[2] = wmma_f16(A0, B2, acc[2]); acc[3] = wmma_f16(A0, B3, acc[3]);       \
  acc[4] = wmma_f16(A1, B0, acc[4]); acc[5] = wmma_f16(A1, B1, acc[5]);       \
  acc[6] = wmma_f16(A1, B2, acc[6]); acc[7] = wmma_f16(A1, B3, acc[7]);

// ---------------------------------------------------------------------------
// WMMA GEMM (f16 in, f32 out):  C[M,N] (+)= rowscale[m] * A[M,K] * Bt[N,K]^T
//   A  : f16 row-major, stride lda (halves), K contiguous
//   Bt : f16 row-major [N,K], stride ldb — B's columns are Bt's rows
//   C  : f32 row-major, stride ldc
// One wave per 32x64 tile; K loop processes TWO 32-wide K steps per
// iteration with ping-pong fragment buffers: buffers are only ever written
// by fresh global loads (never reg-to-reg rotation), so there are no v_mov
// copies and every WMMA group has a full load batch in flight ahead of it.
// ksteps = K/32 is even for every GEMM here (K in {64, 512, 1024}).
// C/D layout: VGPR v -> row (v + (lane>=16 ? 8:0)), col = lane&15.
// M%32==0, N%64==0 hold for every GEMM here (no tails).
// Tile-OOB early return is wave-uniform => EXEC all-ones for WMMA.
// ---------------------------------------------------------------------------
template <bool ACC, bool ROWSCALE>
__global__ __launch_bounds__(256) void wmma_gemm_f16(
    const _Float16* __restrict__ A, int lda,
    const _Float16* __restrict__ Bt, int ldb,
    float* __restrict__ C, int ldc,
    int M, int N, int K,
    const float* __restrict__ rs, int rsStride) {
  int wave   = blockIdx.x * (blockDim.x >> 5) + (threadIdx.x >> 5);
  int tilesN = N >> 6;
  int tiles  = (M >> 5) * tilesN;
  if (wave >= tiles) return;                  // wave-uniform exit
  int m0 = (wave / tilesN) << 5;
  int n0 = (wave % tilesN) << 6;

  int  lane = threadIdx.x & 31;
  int  r15  = lane & 15;
  bool hi   = lane >= 16;

  const _Float16* ap0 = A + (size_t)(m0 + r15) * lda + (hi ? 16 : 0);
  const _Float16* ap1 = ap0 + (size_t)16 * lda;
  const _Float16* bp0 = Bt + (size_t)(n0 + r15) * ldb + (hi ? 16 : 0);
  const _Float16* bp1 = bp0 + (size_t)16 * ldb;
  const _Float16* bp2 = bp0 + (size_t)32 * ldb;
  const _Float16* bp3 = bp0 + (size_t)48 * ldb;

  v8f acc[8] = {};   // [row-group 0/1][4 col-groups]

  // prologue: buffer A <- fragments for k = 0
  v16h aA0 = load_a_frag(ap0);
  v16h aA1 = load_a_frag(ap1);
  v16h bA0 = load_b_frag(bp0);
  v16h bA1 = load_b_frag(bp1);
  v16h bA2 = load_b_frag(bp2);
  v16h bA3 = load_b_frag(bp3);

  int pairs = (K >> 6) - 1;   // iterations over K-step pairs, minus tail pair
#pragma unroll 1
  for (int p = 0; p < pairs; p++) {
    int k1 = (p << 6) + 32;   // buffer B step
    int k2 = k1 + 32;         // buffer A reload step
    // pull the NEXT pair's B-buffer K-tile toward the WGP
    if (k2 + 32 < K) {
      __builtin_prefetch(bp0 + k2 + 32, 0, 1);
      __builtin_prefetch(bp1 + k2 + 32, 0, 1);
      __builtin_prefetch(bp2 + k2 + 32, 0, 1);
      __builtin_prefetch(bp3 + k2 + 32, 0, 1);
    }
    // load buffer B (k1) while WMMAs consume buffer A (k1-32)
    v16h aB0 = load_a_frag(ap0 + k1);
    v16h aB1 = load_a_frag(ap1 + k1);
    v16h bB0 = load_b_frag(bp0 + k1);
    v16h bB1 = load_b_frag(bp1 + k1);
    v16h bB2 = load_b_frag(bp2 + k1);
    v16h bB3 = load_b_frag(bp3 + k1);
    WMMA8(aA0, aA1, bA0, bA1, bA2, bA3)
    // reload buffer A (k2) while WMMAs consume buffer B (k1)
    aA0 = load_a_frag(ap0 + k2);
    aA1 = load_a_frag(ap1 + k2);
    bA0 = load_b_frag(bp0 + k2);
    bA1 = load_b_frag(bp1 + k2);
    bA2 = load_b_frag(bp2 + k2);
    bA3 = load_b_frag(bp3 + k2);
    WMMA8(aB0, aB1, bB0, bB1, bB2, bB3)
  }
  // tail pair: buffer A holds K-64; load buffer B for K-32
  {
    int k1 = K - 32;
    v16h aB0 = load_a_frag(ap0 + k1);
    v16h aB1 = load_a_frag(ap1 + k1);
    v16h bB0 = load_b_frag(bp0 + k1);
    v16h bB1 = load_b_frag(bp1 + k1);
    v16h bB2 = load_b_frag(bp2 + k1);
    v16h bB3 = load_b_frag(bp3 + k1);
    WMMA8(aA0, aA1, bA0, bA1, bA2, bA3)
    WMMA8(aB0, aB1, bB0, bB1, bB2, bB3)
  }

  // store: C/D layout -> global f32
  int rbase = m0 + (hi ? 8 : 0);
#pragma unroll
  for (int g = 0; g < 2; g++) {
#pragma unroll
    for (int j = 0; j < 4; j++) {
      int col = n0 + j * 16 + r15;
#pragma unroll
      for (int v = 0; v < 8; v++) {
        int   r   = rbase + g * 16 + v;
        float val = acc[g * 4 + j][v];
        if (ROWSCALE) val *= rs[(size_t)r * rsStride];
        float* cp = C + (size_t)r * ldc + col;
        if (ACC) *cp += val; else *cp = val;
      }
    }
  }
}

static inline int gemm_blocks(int M, int N) {
  return ((M >> 5) * (N >> 6) + 7) / 8;   // 8 waves (256 thr) per block
}
static inline int cvt_blocks(size_t n) { return (int)((n / 8 + 255) / 256); }

// ---------------------------------------------------------------------------
extern "C" void kernel_launch(void* const* d_in, const int* in_sizes, int n_in,
                              void* d_out, int out_size, void* d_ws, size_t ws_size,
                              hipStream_t stream) {
  (void)in_sizes; (void)n_in; (void)out_size; (void)ws_size;

  const int*   ids          = (const int*)d_in[0];
  const float* tok_emb      = (const float*)d_in[1];
  const float* pos_emb      = (const float*)d_in[2];
  const float* attn_norm_w  = (const float*)d_in[3];
  const float* q_w          = (const float*)d_in[4];
  const float* k_w          = (const float*)d_in[5];
  const float* v_w          = (const float*)d_in[6];
  const float* o_w          = (const float*)d_in[7];
  const float* ffn_norm_w   = (const float*)d_in[8];
  const float* gate_w       = (const float*)d_in[9];
  const float* expert_w     = (const float*)d_in[10];
  const float* final_norm_w = (const float*)d_in[11];
  const float* out_w        = (const float*)d_in[12];
  float*       out          = (float*)d_out;

  // -------- workspace layout (~76 MB, fully rewritten each call) ----------
  float* wsf = (float*)d_ws;
  size_t o = 0;
  float* h      = wsf + o; o += (size_t)NN * CC;          // residual stream
  float* xn     = wsf + o; o += (size_t)NN * CC;          // normed acts (f32)
  float* q      = wsf + o; o += (size_t)NN * CC;          // Q f32
  float* kb     = wsf + o; o += (size_t)NN * KVH * HDIM;  // K f32
  float* vb     = wsf + o; o += (size_t)NN * KVH * HDIM;  // V f32
  float* attn   = wsf + o; o += (size_t)NN * CC;          // attn out f32
  float* scores = wsf + o; o += (size_t)TT * TT;          // per-(b,h) scores
  float* comb   = wsf + o; o += (size_t)NN * EE;          // MoE combine wts

  _Float16* wsh = (_Float16*)(wsf + o);
  size_t oh = 0;
  _Float16* xn_h   = wsh + oh; oh += (size_t)NN * CC;
  _Float16* q_h    = wsh + oh; oh += (size_t)NN * CC;
  _Float16* k_h    = wsh + oh; oh += (size_t)NN * KVH * HDIM;
  _Float16* vT     = wsh + oh; oh += (size_t)BB * KVH * HDIM * TT;
  _Float16* attn_h = wsh + oh; oh += (size_t)NN * CC;
  _Float16* ph     = wsh + oh; oh += (size_t)TT * TT;     // probabilities f16
  _Float16* qw_h   = wsh + oh; oh += (size_t)LL * CC * CC;
  _Float16* kw_h   = wsh + oh; oh += (size_t)LL * KVH * HDIM * CC;
  _Float16* vw_h   = wsh + oh; oh += (size_t)LL * KVH * HDIM * CC;
  _Float16* ow_h   = wsh + oh; oh += (size_t)LL * CC * CC;
  _Float16* expw_h = wsh + oh; oh += (size_t)LL * EE * CC * CC;
  _Float16* outw_h = wsh + oh; oh += (size_t)VV * CC;

  // -------- one-time (per call) f16 weight mirrors (L2-resident) ----------
  cvt_kernel<<<cvt_blocks((size_t)LL * CC * CC), 256, 0, stream>>>(
      q_w, qw_h, (int)((size_t)LL * CC * CC / 8));
  cvt_kernel<<<cvt_blocks((size_t)LL * KVH * HDIM * CC), 256, 0, stream>>>(
      k_w, kw_h, (int)((size_t)LL * KVH * HDIM * CC / 8));
  cvt_kernel<<<cvt_blocks((size_t)LL * KVH * HDIM * CC), 256, 0, stream>>>(
      v_w, vw_h, (int)((size_t)LL * KVH * HDIM * CC / 8));
  cvt_kernel<<<cvt_blocks((size_t)LL * CC * CC), 256, 0, stream>>>(
      o_w, ow_h, (int)((size_t)LL * CC * CC / 8));
  cvt_kernel<<<cvt_blocks((size_t)LL * EE * CC * CC), 256, 0, stream>>>(
      expert_w, expw_h, (int)((size_t)LL * EE * CC * CC / 8));
  cvt_kernel<<<cvt_blocks((size_t)VV * CC), 256, 0, stream>>>(
      out_w, outw_h, (int)((size_t)VV * CC / 8));

  embed_kernel<<<(NN * CC + 255) / 256, 256, 0, stream>>>(ids, tok_emb, pos_emb, h);

  for (int l = 0; l < LL; l++) {
    // ---- attention block ----
    rmsnorm_kernel<<<NN, 256, 0, stream>>>(h, attn_norm_w + l * CC, xn);
    cvt_kernel<<<cvt_blocks((size_t)NN * CC), 256, 0, stream>>>(
        xn, xn_h, NN * CC / 8);

    wmma_gemm_f16<false, false><<<gemm_blocks(NN, CC), 256, 0, stream>>>(
        xn_h, CC, qw_h + (size_t)l * CC * CC, CC, q, CC, NN, CC, CC, nullptr, 0);
    wmma_gemm_f16<false, false><<<gemm_blocks(NN, KVH * HDIM), 256, 0, stream>>>(
        xn_h, CC, kw_h + (size_t)l * KVH * HDIM * CC, CC, kb, KVH * HDIM,
        NN, KVH * HDIM, CC, nullptr, 0);
    wmma_gemm_f16<false, false><<<gemm_blocks(NN, KVH * HDIM), 256, 0, stream>>>(
        xn_h, CC, vw_h + (size_t)l * KVH * HDIM * CC, CC, vb, KVH * HDIM,
        NN, KVH * HDIM, CC, nullptr, 0);

    cvt_kernel<<<cvt_blocks((size_t)NN * CC), 256, 0, stream>>>(q, q_h, NN * CC / 8);
    cvt_kernel<<<cvt_blocks((size_t)NN * KVH * HDIM), 256, 0, stream>>>(
        kb, k_h, NN * KVH * HDIM / 8);
    vtrans_kernel<<<(BB * KVH * HDIM * TT + 255) / 256, 256, 0, stream>>>(vb, vT);

    for (int b = 0; b < BB; b++) {
      for (int hh = 0; hh < HH; hh++) {
        int kv = hh >> 2;   // GQA repeat_interleave: kv head = h / (H/KV)
        const _Float16* qbh = q_h + (size_t)b * TT * CC + hh * HDIM;
        const _Float16* kbh = k_h + (size_t)b * TT * (KVH * HDIM) + kv * HDIM;
        const _Float16* vTb = vT + ((size_t)b * KVH + kv) * HDIM * TT;
        // scores[T,T] = Q_bh [T,HD] @ K_bh[T,HD]^T   (K-contraction = 64)
        wmma_gemm_f16<false, false><<<gemm_blocks(TT, TT), 256, 0, stream>>>(
            qbh, CC, kbh, KVH * HDIM, scores, TT, TT, TT, HDIM, nullptr, 0);
        attn_softmax_kernel<<<TT, 256, 0, stream>>>(scores, ph);
        // attn_bh[T,HD] = P[T,T] @ V_bh[T,HD]; vT rows are V columns (contig K)
        wmma_gemm_f16<false, false><<<gemm_blocks(TT, HDIM), 256, 0, stream>>>(
            ph, TT, vTb, TT, attn + (size_t)b * TT * CC + hh * HDIM, CC,
            TT, HDIM, TT, nullptr, 0);
      }
    }
    // residual: h += attn @ o_w^T
    cvt_kernel<<<cvt_blocks((size_t)NN * CC), 256, 0, stream>>>(
        attn, attn_h, NN * CC / 8);
    wmma_gemm_f16<true, false><<<gemm_blocks(NN, CC), 256, 0, stream>>>(
        attn_h, CC, ow_h + (size_t)l * CC * CC, CC, h, CC, NN, CC, CC, nullptr, 0);

    // ---- MoE block ----
    rmsnorm_kernel<<<NN, 256, 0, stream>>>(h, ffn_norm_w + l * CC, xn);
    gate_top2_kernel<<<NN, 256, 0, stream>>>(xn, gate_w + (size_t)l * EE * CC, comb);
    cvt_kernel<<<cvt_blocks((size_t)NN * CC), 256, 0, stream>>>(
        xn, xn_h, NN * CC / 8);
    for (int e = 0; e < EE; e++) {
      // h[n,:] += comb[n,e] * (xn @ W_e^T)[n,:]   (dense, matching reference)
      wmma_gemm_f16<true, true><<<gemm_blocks(NN, CC), 256, 0, stream>>>(
          xn_h, CC, expw_h + ((size_t)l * EE + e) * CC * CC, CC, h, CC,
          NN, CC, CC, comb + e, EE);
    }
  }

  // ---- final norm + logits (the 67 GF GEMM) ----
  rmsnorm_kernel<<<NN, 256, 0, stream>>>(h, final_norm_w, xn);
  cvt_kernel<<<cvt_blocks((size_t)NN * CC), 256, 0, stream>>>(xn, xn_h, NN * CC / 8);
  wmma_gemm_f16<false, false><<<gemm_blocks(NN, VV), 256, 0, stream>>>(
      xn_h, CC, outw_h, CC, out, VV, NN, VV, CC, nullptr, 0);
}
